// BaseX2HAttLayer_1692217115459
// MI455X (gfx1250) — compile-verified, matrix operand order checked by reading, not run on hardware
//
#include <hip/hip_runtime.h>
#include <hip/hip_bf16.h>

typedef __attribute__((ext_vector_type(16))) __bf16 bf16x16;
typedef __attribute__((ext_vector_type(8)))  float  f32x8;
typedef __attribute__((ext_vector_type(4)))  unsigned int u32x4;

#define D      128
#define NH     16
#define DH     8
#define KV_IN  324
#define KV_PAD 352   // 11 * 32

__device__ __forceinline__ unsigned short f2bf(float x) {
  unsigned int b = __float_as_uint(x);
  b += 0x7FFFu + ((b >> 16) & 1u);          // round-to-nearest-even
  return (unsigned short)(b >> 16);
}
__device__ __forceinline__ float bf2f(unsigned short u) {
  return __uint_as_float(((unsigned int)u) << 16);
}
__device__ __forceinline__ uint2 f2bf4(float4 x) {
  uint2 r;
  r.x = ((unsigned)f2bf(x.x)) | (((unsigned)f2bf(x.y)) << 16);
  r.y = ((unsigned)f2bf(x.z)) | (((unsigned)f2bf(x.w)) << 16);
  return r;
}
// monotone float<->uint mapping for atomicMax on floats
__device__ __forceinline__ unsigned int fmap(float x) {
  unsigned int b = __float_as_uint(x);
  return (b & 0x80000000u) ? ~b : (b | 0x80000000u);
}
__device__ __forceinline__ float funmap(unsigned int m) {
  unsigned int b = (m & 0x80000000u) ? (m & 0x7FFFFFFFu) : ~m;
  return __uint_as_float(b);
}

union FragU { bf16x16 v; u32x4 q[2]; };

// Fragment loader for V_WMMA_F32_16X16X32_BF16 (A from LDS, B from global).
// 16-bit layout: lane&15 = M (A) or N (B); lanes 0-15 hold K {0..7,16..23},
// lanes 16-31 hold K {8..15,24..31} -> two contiguous 16B loads per fragment.
__device__ __forceinline__ bf16x16 load_frag(const unsigned short* base, int ld,
                                             int rcOff, int kOff, int lane) {
  int rc = lane & 15;
  int kb = (lane & 16) ? 8 : 0;
  const unsigned short* p = base + (size_t)(rcOff + rc) * ld + kOff + kb;
  FragU f;
  f.q[0] = *(const u32x4*)(p);
  f.q[1] = *(const u32x4*)(p + 16);
  return f.v;
}

#define WMMA_BF16(A, B, C) \
  __builtin_amdgcn_wmma_f32_16x16x32_bf16(false, (A), false, (B), (short)0, (C), false, false)

// One 32-row tile of: Linear(K->128) -> LayerNorm -> ReLU -> Linear(128->128).
// Block = 128 threads = 4 waves; wave w owns output column tiles {w, w+4};
// each wave processes two 16-row WMMA tiles per B-fragment load (weight reuse x2).
template<int KA>
__device__ __forceinline__ void mlp32(
    const unsigned short* sA,                     // [32][KA] bf16
    const unsigned short* W1T, const float* b1,
    const float* g, const float* beta,
    const unsigned short* W2T, const float* b2,
    float* sHid, unsigned short* sH16, float* sOut,  // [32][128]
    int tid)
{
  const int wave = tid >> 5;
  const int lane = tid & 31;
  const int nn = lane & 15;
  const int mo = (lane & 16) ? 8 : 0;

  for (int c = wave; c < 8; c += 4) {
    f32x8 acc0 = {}, acc1 = {};
#pragma unroll
    for (int k = 0; k < KA; k += 32) {
      bf16x16 b  = load_frag(W1T, KA, c * 16, k, lane);
      bf16x16 a0 = load_frag(sA,  KA, 0,      k, lane);
      bf16x16 a1 = load_frag(sA,  KA, 16,     k, lane);
      acc0 = WMMA_BF16(a0, b, acc0);
      acc1 = WMMA_BF16(a1, b, acc1);
    }
    float bias = b1[c * 16 + nn];
#pragma unroll
    for (int i = 0; i < 8; i++) {
      sHid[(mo + i) * D + c * 16 + nn]      = acc0[i] + bias;
      sHid[(16 + mo + i) * D + c * 16 + nn] = acc1[i] + bias;
    }
  }
  __syncthreads();

  // LayerNorm + ReLU + bf16 pack: 4 threads per row, 32 elements each,
  // reduce across the 4-lane group with shuffles (wave32: never crosses wave).
  {
    const int r = tid >> 2;            // row 0..31
    const int part = tid & 3;          // 32-elem chunk
    const float* row = sHid + r * D + part * 32;
    float s = 0.f, s2 = 0.f;
#pragma unroll
    for (int j = 0; j < 32; j++) { float x = row[j]; s += x; s2 += x * x; }
    s  += __shfl_xor(s, 1);  s2 += __shfl_xor(s2, 1);
    s  += __shfl_xor(s, 2);  s2 += __shfl_xor(s2, 2);
    float mu  = s * (1.f / D);
    float var = fmaxf(s2 * (1.f / D) - mu * mu, 0.f);
    float inv = rsqrtf(var + 1e-5f);
    unsigned short* dst = sH16 + r * D + part * 32;
#pragma unroll
    for (int j = 0; j < 32; j += 4) {
      float4 y;
      y.x = fmaxf((row[j + 0] - mu) * inv * g[part * 32 + j + 0] + beta[part * 32 + j + 0], 0.f);
      y.y = fmaxf((row[j + 1] - mu) * inv * g[part * 32 + j + 1] + beta[part * 32 + j + 1], 0.f);
      y.z = fmaxf((row[j + 2] - mu) * inv * g[part * 32 + j + 2] + beta[part * 32 + j + 2], 0.f);
      y.w = fmaxf((row[j + 3] - mu) * inv * g[part * 32 + j + 3] + beta[part * 32 + j + 3], 0.f);
      *(uint2*)(dst + j) = f2bf4(y);
    }
  }
  __syncthreads();

  for (int c = wave; c < 8; c += 4) {
    f32x8 acc0 = {}, acc1 = {};
#pragma unroll
    for (int k = 0; k < D; k += 32) {
      bf16x16 b  = load_frag(W2T,  D, c * 16, k, lane);
      bf16x16 a0 = load_frag(sH16, D, 0,      k, lane);
      bf16x16 a1 = load_frag(sH16, D, 16,     k, lane);
      acc0 = WMMA_BF16(a0, b, acc0);
      acc1 = WMMA_BF16(a1, b, acc1);
    }
    float bias = b2[c * 16 + nn];
#pragma unroll
    for (int i = 0; i < 8; i++) {
      sOut[(mo + i) * D + c * 16 + nn]      = acc0[i] + bias;
      sOut[(16 + mo + i) * D + c * 16 + nn] = acc1[i] + bias;
    }
  }
  __syncthreads();
}

// ---------------- kernel 1: convert + transpose weights to bf16 ----------------
__global__ void __launch_bounds__(256)
prep_weights(const float* hk_w1, const float* hv_w1, const float* hq_w1,
             const float* hk_w2, const float* hv_w2, const float* hq_w2,
             const float* no_w1, const float* no_w2,
             unsigned short* WkT1, unsigned short* WvT1, unsigned short* WqT1,
             unsigned short* WkT2, unsigned short* WvT2, unsigned short* WqT2,
             unsigned short* WnT1, unsigned short* WnT2)
{
  int i = blockIdx.x * 256 + threadIdx.x;
  if (i < 128 * KV_PAD) {                // [n][k] transposed, K zero-padded
    int n = i / KV_PAD, k = i - n * KV_PAD;
    WkT1[i] = f2bf(k < KV_IN ? hk_w1[k * D + n] : 0.f);
    WvT1[i] = f2bf(k < KV_IN ? hv_w1[k * D + n] : 0.f);
  }
  if (i < 128 * 256) {
    int n = i >> 8, k = i & 255;
    WnT1[i] = f2bf(no_w1[k * D + n]);
  }
  if (i < 128 * 128) {
    int n = i >> 7, k = i & 127;
    WqT1[i] = f2bf(hq_w1[k * D + n]);
    WkT2[i] = f2bf(hk_w2[k * D + n]);
    WvT2[i] = f2bf(hv_w2[k * D + n]);
    WqT2[i] = f2bf(hq_w2[k * D + n]);
    WnT2[i] = f2bf(no_w2[k * D + n]);
  }
}

// ---------------- kernel 2: zero accumulators ----------------
__global__ void __launch_bounds__(256)
init_accum(unsigned int* segmax, float* denom, float* num, int n16, int n128)
{
  int i = blockIdx.x * 256 + threadIdx.x;
  if (i < n16)  { segmax[i] = 0u; denom[i] = 0.f; }
  if (i < n128) num[i] = 0.f;
}

// ---------------- kernel 3: q = MLP_q(h) ----------------
__global__ void __launch_bounds__(128)
node_q_kernel(const float* __restrict__ h,
              const unsigned short* W1T, const float* b1, const float* g, const float* beta,
              const unsigned short* W2T, const float* b2,
              float* __restrict__ qbuf, int N)
{
  __shared__ __align__(16) unsigned short sA[32 * D];
  __shared__ __align__(16) float          sHid[32 * D];
  __shared__ __align__(16) unsigned short sH16[32 * D];
  __shared__ __align__(16) float          sOut[32 * D];
  const int tid = threadIdx.x;
  const int n0 = blockIdx.x * 32;
  for (int i = tid; i < 32 * (D / 4); i += 128) {
    int r = i >> 5, c = (i & 31) * 4, node = n0 + r;
    float4 x = make_float4(0.f, 0.f, 0.f, 0.f);
    if (node < N) x = *(const float4*)(h + (size_t)node * D + c);
    *(uint2*)(sA + r * D + c) = f2bf4(x);
  }
  __syncthreads();
  mlp32<D>(sA, W1T, b1, g, beta, W2T, b2, sHid, sH16, sOut, tid);
  for (int i = tid; i < 32 * (D / 4); i += 128) {
    int r = i >> 5, c = (i & 31) * 4, node = n0 + r;
    if (node < N)
      *(float4*)(qbuf + (size_t)node * D + c) = *(const float4*)(sOut + r * D + c);
  }
}

// ---------------- kernel 4: per-edge k/v MLPs + logits + segment max ----------------
__global__ void __launch_bounds__(128)
edge_kv_kernel(const float* __restrict__ h, const float* __restrict__ r_feat,
               const float* __restrict__ edge_feat, const int* __restrict__ ei,
               const unsigned short* WkT1, const float* kb1, const float* kg, const float* kbe,
               const unsigned short* WkT2, const float* kb2,
               const unsigned short* WvT1, const float* vb1, const float* vg, const float* vbe,
               const unsigned short* WvT2, const float* vb2,
               const float* ew_w, const float* ew_b,
               const float* __restrict__ qbuf,
               float* __restrict__ logits, unsigned int* __restrict__ segmax,
               unsigned short* __restrict__ vbuf, int E)
{
  __shared__ __align__(16) unsigned short sA[32 * KV_PAD];
  __shared__ __align__(16) float          sHid[32 * D];
  __shared__ __align__(16) unsigned short sH16[32 * D];
  __shared__ __align__(16) float          sOut[32 * D];
  __shared__ int   sSrc[32], sDst[32];
  __shared__ float sEw[32];

  const int tid = threadIdx.x;
  const int e0 = blockIdx.x * 32;

  if (tid < 32) {
    int e = min(e0 + tid, E - 1);
    sSrc[tid] = ei[e];
    sDst[tid] = ei[E + e];
  }
  __syncthreads();

  // edge-weight gate: sigmoid(r_feat @ ew_w + ew_b); 4 threads per edge
  {
    const int r = tid >> 2, part = tid & 3;
    int e = min(e0 + r, E - 1);
    float acc = 0.f;
#pragma unroll
    for (int j = 0; j < 16; j++)
      acc += r_feat[(size_t)e * 64 + part * 16 + j] * ew_w[part * 16 + j];
    acc += __shfl_xor(acc, 1);
    acc += __shfl_xor(acc, 2);
    if (part == 0) sEw[r] = 1.f / (1.f + __expf(-(acc + ew_b[0])));
  }

  // kv_input = [edge_feat(4) | r_feat(64) | h[dst](128) | h[src](128) | pad]
  // all segment boundaries are multiples of 4 -> float4 loads, 8B LDS stores
  for (int i = tid; i < 32 * (KV_PAD / 4); i += 128) {
    int r = i / (KV_PAD / 4);
    int kk = (i - r * (KV_PAD / 4)) * 4;
    int e = e0 + r;
    float4 x = make_float4(0.f, 0.f, 0.f, 0.f);
    if (e < E) {
      if      (kk < 4)   x = *(const float4*)(edge_feat + (size_t)e * 4);
      else if (kk < 68)  x = *(const float4*)(r_feat + (size_t)e * 64 + (kk - 4));
      else if (kk < 196) x = *(const float4*)(h + (size_t)sDst[r] * D + (kk - 68));
      else if (kk < 324) x = *(const float4*)(h + (size_t)sSrc[r] * D + (kk - 196));
    }
    *(uint2*)(sA + r * KV_PAD + kk) = f2bf4(x);
  }
  __syncthreads();

  // k = MLP_k(kv_input); logits = (q[dst] . k) / sqrt(dh); running segment max
  mlp32<KV_PAD>(sA, WkT1, kb1, kg, kbe, WkT2, kb2, sHid, sH16, sOut, tid);
  for (int t = tid; t < 512; t += 128) {
    int r = t >> 4, hd = t & 15;
    int e = e0 + r;
    if (e < E) {
      int dn = sDst[r];
      float s = 0.f;
#pragma unroll
      for (int j = 0; j < DH; j++)
        s += qbuf[(size_t)dn * D + hd * DH + j] * sOut[r * D + hd * DH + j];
      s *= 0.3535533906f;                  // 1/sqrt(8)
      logits[(size_t)e * NH + hd] = s;
      atomicMax(&segmax[dn * NH + hd], fmap(s));
    }
  }

  // v = MLP_v(kv_input) * e_w   (stored bf16, 4-wide packs)
  mlp32<KV_PAD>(sA, WvT1, vb1, vg, vbe, WvT2, vb2, sHid, sH16, sOut, tid);
  for (int i = tid; i < 32 * (D / 4); i += 128) {
    int r = i >> 5, c = (i & 31) * 4;
    int e = e0 + r;
    if (e < E) {
      float w = sEw[r];
      float4 x = *(const float4*)(sOut + r * D + c);
      x.x *= w; x.y *= w; x.z *= w; x.w *= w;
      *(uint2*)(vbuf + (size_t)e * D + c) = f2bf4(x);
    }
  }
}

// ---------------- kernel 5: ex = exp(logit - segmax); accumulate denom & ex*v ----------------
// one thread per (edge, 4-channel chunk); chunks never cross a head boundary
__global__ void __launch_bounds__(256)
scatter_kernel(const int* __restrict__ ei, const float* __restrict__ logits,
               const unsigned int* __restrict__ segmax, const unsigned short* __restrict__ vbuf,
               float* __restrict__ denom, float* __restrict__ num, int E)
{
  int i = blockIdx.x * 256 + threadIdx.x;
  int total = E * (D / 4);                 // 10.24M
  if (i >= total) return;
  int e = i >> 5, c = (i & 31) * 4, hd = c >> 3;
  int dn = ei[E + e];
  float mx = funmap(segmax[dn * NH + hd]);
  float ex = __expf(logits[(size_t)e * NH + hd] - mx);
  if ((c & 7) == 0) atomicAdd(&denom[dn * NH + hd], ex);
  uint2 v = *(const uint2*)(vbuf + (size_t)e * D + c);
  float* dst = num + (size_t)dn * D + c;
  atomicAdd(dst + 0, ex * bf2f((unsigned short)(v.x & 0xFFFFu)));
  atomicAdd(dst + 1, ex * bf2f((unsigned short)(v.x >> 16)));
  atomicAdd(dst + 2, ex * bf2f((unsigned short)(v.y & 0xFFFFu)));
  atomicAdd(dst + 3, ex * bf2f((unsigned short)(v.y >> 16)));
}

// ---------------- kernel 6: out = MLP_no([agg | h]) + h ----------------
__global__ void __launch_bounds__(128)
node_out_kernel(const float* __restrict__ h, const float* __restrict__ num,
                const float* __restrict__ denom,
                const unsigned short* W1T, const float* b1, const float* g, const float* beta,
                const unsigned short* W2T, const float* b2,
                float* __restrict__ out, int N)
{
  __shared__ __align__(16) unsigned short sA[32 * 256];
  __shared__ __align__(16) float          sHid[32 * D];
  __shared__ __align__(16) unsigned short sH16[32 * D];
  __shared__ __align__(16) float          sOut[32 * D];
  const int tid = threadIdx.x;
  const int n0 = blockIdx.x * 32;
  for (int i = tid; i < 32 * (256 / 4); i += 128) {
    int r = i >> 6, c = (i & 63) * 4;
    int node = n0 + r;
    float4 x = make_float4(0.f, 0.f, 0.f, 0.f);
    if (node < N) {
      if (c < D) {                         // aggregated attention output
        float den = denom[node * NH + (c >> 3)];
        if (den > 0.f) {
          float inv = 1.f / den;
          float4 nm = *(const float4*)(num + (size_t)node * D + c);
          x = make_float4(nm.x * inv, nm.y * inv, nm.z * inv, nm.w * inv);
        }
      } else {                             // residual input h
        x = *(const float4*)(h + (size_t)node * D + (c - D));
      }
    }
    *(uint2*)(sA + r * 256 + c) = f2bf4(x);
  }
  __syncthreads();
  mlp32<256>(sA, W1T, b1, g, beta, W2T, b2, sHid, sH16, sOut, tid);
  for (int i = tid; i < 32 * (D / 4); i += 128) {
    int r = i >> 5, c = (i & 31) * 4;
    int node = n0 + r;
    if (node < N) {
      float4 hv = *(const float4*)(h + (size_t)node * D + c);
      float4 y  = *(const float4*)(sOut + r * D + c);
      y.x += hv.x; y.y += hv.y; y.z += hv.z; y.w += hv.w;
      *(float4*)(out + (size_t)node * D + c) = y;
    }
  }
}

extern "C" void kernel_launch(void* const* d_in, const int* in_sizes, int n_in,
                              void* d_out, int out_size, void* d_ws, size_t ws_size,
                              hipStream_t stream) {
  const float* h         = (const float*)d_in[0];
  const float* r_feat    = (const float*)d_in[1];
  const float* edge_feat = (const float*)d_in[2];
  const int*   ei        = (const int*)d_in[3];
  const float* hk_w1 = (const float*)d_in[4];  const float* hk_b1 = (const float*)d_in[5];
  const float* hk_g  = (const float*)d_in[6];  const float* hk_be = (const float*)d_in[7];
  const float* hk_w2 = (const float*)d_in[8];  const float* hk_b2 = (const float*)d_in[9];
  const float* hv_w1 = (const float*)d_in[10]; const float* hv_b1 = (const float*)d_in[11];
  const float* hv_g  = (const float*)d_in[12]; const float* hv_be = (const float*)d_in[13];
  const float* hv_w2 = (const float*)d_in[14]; const float* hv_b2 = (const float*)d_in[15];
  const float* hq_w1 = (const float*)d_in[16]; const float* hq_b1 = (const float*)d_in[17];
  const float* hq_g  = (const float*)d_in[18]; const float* hq_be = (const float*)d_in[19];
  const float* hq_w2 = (const float*)d_in[20]; const float* hq_b2 = (const float*)d_in[21];
  const float* ew_w  = (const float*)d_in[22]; const float* ew_b  = (const float*)d_in[23];
  const float* no_w1 = (const float*)d_in[24]; const float* no_b1 = (const float*)d_in[25];
  const float* no_g  = (const float*)d_in[26]; const float* no_be = (const float*)d_in[27];
  const float* no_w2 = (const float*)d_in[28]; const float* no_b2 = (const float*)d_in[29];

  const int N = in_sizes[0] / D;   // 10000
  const int E = in_sizes[3] / 2;   // 320000

  // workspace layout (all offsets 256B aligned)
  char* ws = (char*)d_ws;
  unsigned short* WkT1 = (unsigned short*)(ws + 0);           // 128*352 bf16
  unsigned short* WvT1 = (unsigned short*)(ws + 90112);
  unsigned short* WqT1 = (unsigned short*)(ws + 180224);      // 128*128
  unsigned short* WkT2 = (unsigned short*)(ws + 212992);
  unsigned short* WvT2 = (unsigned short*)(ws + 245760);
  unsigned short* WqT2 = (unsigned short*)(ws + 278528);
  unsigned short* WnT1 = (unsigned short*)(ws + 311296);      // 128*256
  unsigned short* WnT2 = (unsigned short*)(ws + 376832);
  float*          qbuf   = (float*)(ws + 409600);             // N*128 f32
  size_t o = 409600 + (size_t)N * D * 4;
  unsigned int*   segmax = (unsigned int*)(ws + o); o += (size_t)N * NH * 4;
  float*          denom  = (float*)(ws + o);        o += (size_t)N * NH * 4;
  float*          num    = (float*)(ws + o);        o += (size_t)N * D * 4;
  float*          logits = (float*)(ws + o);        o += (size_t)E * NH * 4;
  unsigned short* vbuf   = (unsigned short*)(ws + o);         // E*128 bf16

  prep_weights<<<(128 * KV_PAD + 255) / 256, 256, 0, stream>>>(
      hk_w1, hv_w1, hq_w1, hk_w2, hv_w2, hq_w2, no_w1, no_w2,
      WkT1, WvT1, WqT1, WkT2, WvT2, WqT2, WnT1, WnT2);

  init_accum<<<(N * D + 255) / 256, 256, 0, stream>>>(segmax, denom, num, N * NH, N * D);

  node_q_kernel<<<(N + 31) / 32, 128, 0, stream>>>(
      h, WqT1, hq_b1, hq_g, hq_be, WqT2, hq_b2, qbuf, N);

  edge_kv_kernel<<<(E + 31) / 32, 128, 0, stream>>>(
      h, r_feat, edge_feat, ei,
      WkT1, hk_b1, hk_g, hk_be, WkT2, hk_b2,
      WvT1, hv_b1, hv_g, hv_be, WvT2, hv_b2,
      ew_w, ew_b, qbuf, logits, segmax, vbuf, E);

  scatter_kernel<<<(E * (D / 4) + 255) / 256, 256, 0, stream>>>(
      ei, logits, segmax, vbuf, denom, num, E);

  node_out_kernel<<<(N + 31) / 32, 128, 0, stream>>>(
      h, num, denom, WnT1, no_b1, no_g, no_be, WnT2, no_b2,
      (float*)d_out, N);
}
